// SelfAttention_36850819400191
// MI455X (gfx1250) — matrix operand order repaired and checked by reading, not compiled, not run on hardware
//
#include <hip/hip_runtime.h>
#include <hip/hip_bf16.h>

typedef __attribute__((ext_vector_type(16))) __bf16 v16bf;
typedef __attribute__((ext_vector_type(8)))  float  v8f;
typedef __attribute__((ext_vector_type(4)))  unsigned short u16x4;
typedef __attribute__((ext_vector_type(8)))  unsigned short u16x8;

#define GEMM_BM 128
#define GEMM_BN 128
#define GEMM_BK 64
#define AS2     72    // LDS row stride for 64-wide bf16 tiles: 144B, 16B-aligned, conflict-free
#define ATT_CK  64    // keys per attention chunk
#define ATT_VTS 72
#define ATT_PS  72

union FragBF { v16bf v; uint4 q[2]; };

__device__ __forceinline__ unsigned short f2bf(float f) {
    unsigned int x = __float_as_uint(f);
    unsigned int r = x + 0x7FFFu + ((x >> 16) & 1u);   // round-to-nearest-even
    return (unsigned short)(r >> 16);
}

__device__ __forceinline__ v8f wmma_bf16(const FragBF& a, const FragBF& b, v8f c) {
    return __builtin_amdgcn_wmma_f32_16x16x32_bf16(false, a.v, false, b.v,
                                                   (short)0, c, false, false);
}

__device__ __forceinline__ u16x4 cvt4(const float* p) {
    const float4 f = *(const float4*)p;
    u16x4 u;
    u[0] = f2bf(f.x); u[1] = f2bf(f.y); u[2] = f2bf(f.z); u[3] = f2bf(f.w);
    return u;
}

__device__ __forceinline__ void store_out(float* p, float x) { *p = x; }
__device__ __forceinline__ void store_out(unsigned short* p, float x) { *p = f2bf(x); }

// ======================================================================
// fp32 -> bf16 bulk conversion (one-time pass for x's and W's)
// ======================================================================
__global__ __launch_bounds__(256) void cvt_f32_bf16(
    const float* __restrict__ in, unsigned short* __restrict__ out, long n4) {
    long i = (long)blockIdx.x * blockDim.x + threadIdx.x;
    const long stride = (long)gridDim.x * blockDim.x;
    for (; i < n4; i += stride)
        *(u16x4*)&out[i * 4] = cvt4(&in[i * 4]);
}

// ======================================================================
// int32 mask -> bitmask (1 bit/key). wave32 ballot = exactly one 32-bit word.
// ======================================================================
__global__ __launch_bounds__(256) void pack_mask(
    const int* __restrict__ m, unsigned int* __restrict__ bits, long n) {
    const long t = (long)blockIdx.x * blockDim.x + threadIdx.x;
    const int pred = (t < n) ? (m[t] != 0) : 0;
    const unsigned long long bal = __ballot(pred);
    if ((threadIdx.x & 31) == 0) bits[t >> 5] = (unsigned int)bal;
}

// ======================================================================
// NT GEMM (all-bf16 inputs): C[m][n] = sum_k A[m][k]*Bw[n][k] + bias[n]
// Block tile 128x128, 8 waves (2x4), wave tile 64x32 (4x2 WMMA), BK=64.
// Branchless double-buffered LDS pipeline: next tile's global loads are
// issued before the 16 WMMAs and only waited on at the LDS store after;
// one barrier per K-step. Last iteration does a harmless redundant reload.
// ======================================================================
template <typename TO, bool GELU>
__global__ __launch_bounds__(256) void gemm_nt_wmma(
    const unsigned short* __restrict__ A, const unsigned short* __restrict__ Bw,
    const float* __restrict__ bias, TO* __restrict__ Cout,
    int M, int N, int K) {
    __shared__ __align__(16) unsigned short As[2][GEMM_BM * AS2];
    __shared__ __align__(16) unsigned short Bs[2][GEMM_BN * AS2];

    const int tid  = threadIdx.x;
    const int lane = tid & 31;
    const int w    = tid >> 5;
    const int wm   = w & 1;      // row half    -> 64 rows
    const int wn   = w >> 1;     // col quarter -> 32 cols
    const int r    = lane & 15;
    const int g    = lane >> 4;

    const long rowBase = (long)blockIdx.y * GEMM_BM;
    const long colBase = (long)blockIdx.x * GEMM_BN;

    const v8f vz = {0.f,0.f,0.f,0.f,0.f,0.f,0.f,0.f};
    v8f acc[4][2] = {{vz, vz}, {vz, vz}, {vz, vz}, {vz, vz}};

    const int srow = tid >> 3;        // 0..31
    const int scol = (tid & 7) * 8;   // 0,8,...,56 (16B chunks)
    const unsigned short* Aptr = &A[(rowBase + srow) * (long)K + scol];
    const unsigned short* Bptr = &Bw[(colBase + srow) * (long)K + scol];

    // prologue: stage tile 0
    u16x8 ra[4], rb[4];
    #pragma unroll
    for (int p = 0; p < 4; ++p) {
        ra[p] = *(const u16x8*)(Aptr + (long)(p * 32) * K);
        rb[p] = *(const u16x8*)(Bptr + (long)(p * 32) * K);
    }
    #pragma unroll
    for (int p = 0; p < 4; ++p) {
        *(u16x8*)&As[0][(p * 32 + srow) * AS2 + scol] = ra[p];
        *(u16x8*)&Bs[0][(p * 32 + srow) * AS2 + scol] = rb[p];
    }
    __syncthreads();

    int cur = 0;
    for (int k0 = 0; k0 < K; k0 += GEMM_BK) {
        // next tile index, clamped (last iter reloads the same tile; harmless)
        const int kn = (k0 + GEMM_BK < K) ? (k0 + GEMM_BK) : k0;
        #pragma unroll
        for (int p = 0; p < 4; ++p) {
            ra[p] = *(const u16x8*)(Aptr + (long)(p * 32) * K + kn);
            rb[p] = *(const u16x8*)(Bptr + (long)(p * 32) * K + kn);
        }
        __builtin_prefetch(Aptr + kn + GEMM_BK, 0, 3);   // tile after next
        __builtin_prefetch(Bptr + kn + GEMM_BK, 0, 3);

        const unsigned short* Ac = As[cur];
        const unsigned short* Bc = Bs[cur];
        #pragma unroll
        for (int kk = 0; kk < GEMM_BK; kk += 32) {
            FragBF a[4], b[2];
            #pragma unroll
            for (int i = 0; i < 4; ++i) {   // A layout: K chunks g*8 and 16+g*8
                const int base = (wm * 64 + i * 16 + r) * AS2 + kk;
                a[i].q[0] = *(const uint4*)&Ac[base + g * 8];
                a[i].q[1] = *(const uint4*)&Ac[base + 16 + g * 8];
            }
            #pragma unroll
            for (int j = 0; j < 2; ++j) {   // B layout: K chunks g*8 and 16+g*8
                const int base = (wn * 32 + j * 16 + r) * AS2 + kk;
                b[j].q[0] = *(const uint4*)&Bc[base + g * 8];
                b[j].q[1] = *(const uint4*)&Bc[base + 16 + g * 8];
            }
            #pragma unroll
            for (int i = 0; i < 4; ++i)
                #pragma unroll
                for (int j = 0; j < 2; ++j)
                    acc[i][j] = wmma_bf16(a[i], b[j], acc[i][j]);
        }

        #pragma unroll
        for (int p = 0; p < 4; ++p) {
            *(u16x8*)&As[cur ^ 1][(p * 32 + srow) * AS2 + scol] = ra[p];
            *(u16x8*)&Bs[cur ^ 1][(p * 32 + srow) * AS2 + scol] = rb[p];
        }
        __syncthreads();   // single barrier per K-step
        cur ^= 1;
    }

    // epilogue: bias (+ exact GELU). C layout: row = v + 8*g, col = lane&15
    #pragma unroll
    for (int j = 0; j < 2; ++j) {
        const long col = colBase + wn * 32 + j * 16 + r;
        const float bcol = bias[col];
        #pragma unroll
        for (int i = 0; i < 4; ++i) {
            #pragma unroll
            for (int v = 0; v < 8; ++v) {
                const long row = rowBase + wm * 64 + i * 16 + v + 8 * g;
                float x = acc[i][j][v] + bcol;
                if (GELU) x = 0.5f * x * (1.0f + erff(x * 0.70710678118654752f));
                store_out(&Cout[row * (long)N + col], x);
            }
        }
    }
}

// ======================================================================
// Flash attention: block = (b,h) x 64 queries; 4 waves x 16 q-rows.
// D=64 in registers; 64-key chunks, online softmax, P.1 row-sum WMMA.
// Branchless double-buffered V^T pipeline. Load issue order per chunk:
//   mask word -> score K frags/WMMAs -> next-chunk V loads -> softmax
//   -> PV WMMAs -> VT store (first wait on V loads) -> barrier.
// So V loads stay outstanding through softmax+PV, and score WMMAs never
// wait on them (loads retire in order).
// ======================================================================
__global__ __launch_bounds__(128) void attn_flash(
    const unsigned short* __restrict__ Qb, const unsigned short* __restrict__ Kb,
    const unsigned short* __restrict__ Vb, const unsigned int* __restrict__ mbits,
    unsigned short* __restrict__ Ctx, int S, int E) {
    __shared__ __align__(16) unsigned short VT[2][64 * ATT_VTS];  // V chunk transposed [d][k]
    __shared__ __align__(16) unsigned short Ps[4][16 * ATT_PS];   // per-wave P tile [q][k]

    const int tid = threadIdx.x, lane = tid & 31, w = tid >> 5;
    const int r = lane & 15, g = lane >> 4;
    const int b = blockIdx.z, h = blockIdx.y;
    const int q0 = blockIdx.x * 64 + w * 16;
    const int SW = S >> 5;   // mask words per row

    const long headOff = (long)h * 64;
    const unsigned short* Qh = Qb + (long)b * S * E + headOff;
    const unsigned short* Kh = Kb + (long)b * S * E + headOff;
    const unsigned short* Vh = Vb + (long)b * S * E + headOff;

    // Q fragments (A-operand), reused across all key chunks
    FragBF aq[2];
    #pragma unroll
    for (int dt = 0; dt < 2; ++dt) {
        const unsigned short* p = Qh + (long)(q0 + r) * E + dt * 32;
        aq[dt].q[0] = *(const uint4*)(p + g * 8);
        aq[dt].q[1] = *(const uint4*)(p + 16 + g * 8);
    }

    // all-ones B fragment (bf16 1.0 = 0x3F80) for row-sum WMMA
    FragBF fone;
    fone.q[0] = make_uint4(0x3F803F80u, 0x3F803F80u, 0x3F803F80u, 0x3F803F80u);
    fone.q[1] = fone.q[0];

    const v8f vz = {0.f,0.f,0.f,0.f,0.f,0.f,0.f,0.f};
    v8f o[4] = {vz, vz, vz, vz};
    v8f ol = vz;                  // row-sum accumulator (replicated over 16 cols)
    float mrow[8];
    #pragma unroll
    for (int v = 0; v < 8; ++v) mrow[v] = -3.0e38f;

    const float scale = 0.125f;  // 1/sqrt(64)
    unsigned short* PsW = Ps[w];

    const int vkk = tid >> 4;          // 0..7
    const int vdg = (tid & 15) * 4;    // 0,4,...,60

    // prologue: stage V^T chunk 0
    u16x4 rv[8];
    #pragma unroll
    for (int p = 0; p < 8; ++p)
        rv[p] = *(const u16x4*)(Vh + (long)(p * 8 + vkk) * E + vdg);
    #pragma unroll
    for (int p = 0; p < 8; ++p) {
        const int krow = p * 8 + vkk;
        #pragma unroll
        for (int i = 0; i < 4; ++i)
            VT[0][(vdg + i) * ATT_VTS + krow] = rv[p][i];
    }
    __syncthreads();

    int cur = 0;
    for (int k0 = 0; k0 < S; k0 += ATT_CK) {
        // next chunk base, clamped (last iter reloads same chunk; harmless)
        const int knext = (k0 + ATT_CK < S) ? (k0 + ATT_CK) : k0;

        // (1) mask words for rows q0..q0+15: one lane-parallel b64 load, issued first
        const uint2 mw2 = *(const uint2*)&mbits[((long)b * S + q0 + r) * SW + (k0 >> 5)];

        // (2) scores S = Q . K^T (K rows load directly in B-operand layout)
        v8f s[4];
        #pragma unroll
        for (int nt = 0; nt < 4; ++nt) {
            v8f sacc = vz;
            const unsigned short* kp = Kh + (long)(k0 + nt * 16 + r) * E;
            __builtin_prefetch(kp + (long)ATT_CK * E, 0, 3);
            #pragma unroll
            for (int dt = 0; dt < 2; ++dt) {
                FragBF bk;
                bk.q[0] = *(const uint4*)(kp + dt * 32 + g * 8);
                bk.q[1] = *(const uint4*)(kp + dt * 32 + 16 + g * 8);
                sacc = wmma_bf16(aq[dt], bk, sacc);
            }
            s[nt] = sacc;
        }

        // (3) issue next chunk's V loads; consumed only at the VT store below
        #pragma unroll
        for (int p = 0; p < 8; ++p)
            rv[p] = *(const u16x4*)(Vh + (long)(knext + p * 8 + vkk) * E + vdg);

        // (4) online softmax; row stats per (v,g)
        #pragma unroll
        for (int v = 0; v < 8; ++v) {
            const int src = v + 8 * g;                        // lane holding this row's words
            const unsigned int mlo = (unsigned int)__shfl((int)mw2.x, src, 32);
            const unsigned int mhi = (unsigned int)__shfl((int)mw2.y, src, 32);
            float sv0 = s[0][v] * scale; if (!((mlo >> r)        & 1u)) sv0 = -1e9f;
            float sv1 = s[1][v] * scale; if (!((mlo >> (16 + r)) & 1u)) sv1 = -1e9f;
            float sv2 = s[2][v] * scale; if (!((mhi >> r)        & 1u)) sv2 = -1e9f;
            float sv3 = s[3][v] * scale; if (!((mhi >> (16 + r)) & 1u)) sv3 = -1e9f;
            float rm = fmaxf(fmaxf(sv0, sv1), fmaxf(sv2, sv3));
            rm = fmaxf(rm, __shfl_xor(rm, 1, 16));
            rm = fmaxf(rm, __shfl_xor(rm, 2, 16));
            rm = fmaxf(rm, __shfl_xor(rm, 4, 16));
            rm = fmaxf(rm, __shfl_xor(rm, 8, 16));
            const float mn    = fmaxf(mrow[v], rm);
            const float alpha = __expf(mrow[v] - mn);
            mrow[v] = mn;
            o[0][v] *= alpha; o[1][v] *= alpha; o[2][v] *= alpha; o[3][v] *= alpha;
            ol[v]   *= alpha;
            const int prow = v + 8 * g;
            PsW[prow * ATT_PS + r]      = f2bf(__expf(sv0 - mn));
            PsW[prow * ATT_PS + 16 + r] = f2bf(__expf(sv1 - mn));
            PsW[prow * ATT_PS + 32 + r] = f2bf(__expf(sv2 - mn));
            PsW[prow * ATT_PS + 48 + r] = f2bf(__expf(sv3 - mn));
        }

        // (5) ctx += P.V and l += P.1 (two 32-k steps) from VT[cur]
        const unsigned short* VTc = VT[cur];
        #pragma unroll
        for (int kk = 0; kk < ATT_CK; kk += 32) {
            FragBF pa;
            pa.q[0] = *(const uint4*)&PsW[r * ATT_PS + kk + g * 8];
            pa.q[1] = *(const uint4*)&PsW[r * ATT_PS + kk + 16 + g * 8];
            ol = wmma_bf16(pa, fone, ol);     // row sums (replicated across cols)
            #pragma unroll
            for (int j = 0; j < 4; ++j) {
                FragBF bv;
                const int d = j * 16 + r;
                bv.q[0] = *(const uint4*)&VTc[d * ATT_VTS + kk + g * 8];
                bv.q[1] = *(const uint4*)&VTc[d * ATT_VTS + kk + 16 + g * 8];
                o[j] = wmma_bf16(pa, bv, o[j]);
            }
        }

        // (6) stage next chunk into the other buffer; single barrier per chunk
        #pragma unroll
        for (int p = 0; p < 8; ++p) {
            const int krow = p * 8 + vkk;
            #pragma unroll
            for (int i = 0; i < 4; ++i)
                VT[cur ^ 1][(vdg + i) * ATT_VTS + krow] = rv[p][i];
        }
        __syncthreads();
        cur ^= 1;
    }

    // normalize and write ctx (bf16)
    unsigned short* Ch = Ctx + (long)b * S * E + headOff;
    #pragma unroll
    for (int j = 0; j < 4; ++j) {
        const int col = j * 16 + r;
        #pragma unroll
        for (int v = 0; v < 8; ++v) {
            const int qrow = q0 + v + 8 * g;
            Ch[(long)qrow * E + col] = f2bf(o[j][v] / ol[v]);
        }
    }
}

extern "C" void kernel_launch(void* const* d_in, const int* in_sizes, int n_in,
                              void* d_out, int out_size, void* d_ws, size_t ws_size,
                              hipStream_t stream) {
    const int Bsz = 4, S = 2048, E = 1024, H = 16;
    const float* xv   = (const float*)d_in[0];
    const float* xk   = (const float*)d_in[1];
    const float* xq   = (const float*)d_in[2];
    const int*   mask = (const int*)  d_in[3];
    const float* Wq   = (const float*)d_in[4];
    const float* bq   = (const float*)d_in[5];
    const float* Wk   = (const float*)d_in[6];
    const float* bk   = (const float*)d_in[7];
    const float* Wv   = (const float*)d_in[8];
    const float* bv   = (const float*)d_in[9];
    const float* Wo   = (const float*)d_in[10];
    const float* bo   = (const float*)d_in[11];
    float* out = (float*)d_out;

    const long tok = (long)Bsz * S * E;    // 8M elements
    const long WE  = (long)E * E;          // 1M elements
    unsigned short* base = (unsigned short*)d_ws;
    unsigned short* Xq  = base;
    unsigned short* Xk  = Xq + tok;
    unsigned short* Xv  = Xk + tok;
    unsigned short* Wqb = Xv + tok;
    unsigned short* Wkb = Wqb + WE;
    unsigned short* Wvb = Wkb + WE;
    unsigned short* Wob = Wvb + WE;
    unsigned short* Qb  = Wob + WE;
    unsigned short* Kbp = Qb + tok;
    unsigned short* Vbp = Kbp + tok;
    unsigned short* Ctx = Vbp + tok;
    unsigned int*   mbits = (unsigned int*)(Ctx + tok);   // 2 MB

    // one-time fp32 -> bf16 conversions
    cvt_f32_bf16<<<2048, 256, 0, stream>>>(xq, Xq, tok / 4);
    cvt_f32_bf16<<<2048, 256, 0, stream>>>(xk, Xk, tok / 4);
    cvt_f32_bf16<<<2048, 256, 0, stream>>>(xv, Xv, tok / 4);
    cvt_f32_bf16<<<512,  256, 0, stream>>>(Wq, Wqb, WE / 4);
    cvt_f32_bf16<<<512,  256, 0, stream>>>(Wk, Wkb, WE / 4);
    cvt_f32_bf16<<<512,  256, 0, stream>>>(Wv, Wvb, WE / 4);
    cvt_f32_bf16<<<512,  256, 0, stream>>>(Wo, Wob, WE / 4);

    // pack mask -> 1 bit per key
    const long nmask = (long)Bsz * S * S;              // 16.7M, divisible by 8192
    pack_mask<<<(unsigned)(nmask / 256), 256, 0, stream>>>(mask, mbits, nmask);

    dim3 gg(E / GEMM_BN, (Bsz * S) / GEMM_BM);         // (8, 64)
    gemm_nt_wmma<unsigned short, false><<<gg, 256, 0, stream>>>(Xq, Wqb, bq, Qb,  Bsz * S, E, E);
    gemm_nt_wmma<unsigned short, false><<<gg, 256, 0, stream>>>(Xk, Wkb, bk, Kbp, Bsz * S, E, E);
    gemm_nt_wmma<unsigned short, false><<<gg, 256, 0, stream>>>(Xv, Wvb, bv, Vbp, Bsz * S, E, E);

    dim3 ga(S / 64, H, Bsz);                            // (32, 16, 4)
    attn_flash<<<ga, 128, 0, stream>>>(Qb, Kbp, Vbp, mbits, Ctx, S, E);

    gemm_nt_wmma<float, true><<<gg, 256, 0, stream>>>(Ctx, Wob, bo, out, Bsz * S, E, E);
}
